// EBeehive_3616362463453
// MI455X (gfx1250) — compile-verified, matrix operand order checked by reading, not run on hardware
//
#include <hip/hip_runtime.h>
#include <cstddef>
#include <cstdint>

// ---------------------------------------------------------------------------
// Reflected-diffusion deep-BSDE rollout for MI455X (gfx1250, wave32).
// - 64x64 hidden layers on v_wmma_f32_16x16x32_f16 (f16 in, f32 accum)
// - per-step weights pre-converted to f16 (transposed) once into d_ws,
//   then staged into LDS each step by the Tensor Data Mover (padded rows)
// - tanh via v_exp_f32 + v_rcp_f32; A-fragments explicitly hoisted so each
//   is loaded from LDS once per M-tile (compiler was re-loading them)
// ---------------------------------------------------------------------------

typedef __attribute__((ext_vector_type(16))) _Float16 v16h;
typedef __attribute__((ext_vector_type(8)))  float    v8f;

#define HPAD     72     // row stride in halves: 64 + 8 pad -> 144B (16B aligned, conflict-free)
#define NTHREADS 128
#define WAVES    4

#if defined(__has_builtin)
#if __has_builtin(__builtin_amdgcn_tensor_load_to_lds) && __has_builtin(__builtin_amdgcn_s_wait_tensorcnt)
#define HAVE_TDM 1
#endif
#endif
#ifndef HAVE_TDM
#define HAVE_TDM 0
#endif

struct __align__(16) SMem {
  _Float16 W2t[64 * HPAD];            // f16 weights, transposed [n][k], padded rows
  _Float16 W3t[64 * HPAD];
  _Float16 h[2][WAVES * 32 * HPAD];   // ping-pong activation buffers, per wave
  float    W1f[3 * 64];               // layer-1 weights [k][n], f32
  float    b1[64];
  float    b2[64];
  float    b3[64];
  float    W4f[64 * 4];               // head weights [k][e], padded to 4
  float    b4[4];
};

union AFrag { v16h v; float4 f4[2]; };

__device__ __forceinline__ float fast_tanh(float x) {
  float ax = __builtin_fabsf(x);
  float e  = __expf(-2.0f * ax);
  float t  = (1.0f - e) * __builtin_amdgcn_rcpf(1.0f + e);   // single v_rcp_f32
  return __builtin_copysignf(t, x);
}

// ---- staging helpers -------------------------------------------------------

__device__ __forceinline__ void stage_small(SMem& s, int tid,
    const float* __restrict__ W1, const float* __restrict__ b1_,
    const float* __restrict__ b2_, const float* __restrict__ b3_,
    const float* __restrict__ W4, const float* __restrict__ b4_, int e_out)
{
  for (int i = tid; i < 192; i += NTHREADS) s.W1f[i] = W1[i];
  if (tid < 64) { s.b1[tid] = b1_[tid]; s.b2[tid] = b2_[tid]; s.b3[tid] = b3_[tid]; }
  for (int i = tid; i < 256; i += NTHREADS) {
    int k = i >> 2, e = i & 3;
    s.W4f[i] = (e < e_out) ? W4[k * e_out + e] : 0.0f;
  }
  if (tid < 4) s.b4[tid] = (tid < e_out) ? b4_[tid] : 0.0f;
}

__device__ __forceinline__ void stage_big_f32(SMem& s, int tid,
    const float* __restrict__ W2, const float* __restrict__ W3)
{
  for (int i = tid; i < 4096; i += NTHREADS) {
    int k = i >> 6, n = i & 63;                 // W[k][n] row-major in global
    s.W2t[n * HPAD + k] = (_Float16)W2[i];
    s.W3t[n * HPAD + k] = (_Float16)W3[i];
  }
}

__device__ __forceinline__ void stage_big_f16(SMem& s, int tid,
    const _Float16* __restrict__ w2, const _Float16* __restrict__ w3)
{
  for (int i = tid; i < 4096; i += NTHREADS) {
    int n = i >> 6, k = i & 63;                 // pre-transposed compact [n][k]
    s.W2t[n * HPAD + k] = w2[i];
    s.W3t[n * HPAD + k] = w3[i];
  }
}

// TDM: copy 64x64 f16 tile (compact, row stride 128B) from global into LDS,
// inserting 16B padding after every 128B so LDS rows land on a 144B stride.
__device__ __forceinline__ void tdm_load_tile(uint32_t lds_off, const _Float16* src) {
#if HAVE_TDM
  typedef unsigned int v4u_t __attribute__((ext_vector_type(4)));
  typedef int          v4i_t __attribute__((ext_vector_type(4)));
  typedef int          v8i_t __attribute__((ext_vector_type(8)));
  uint64_t ga = (uint64_t)(uintptr_t)src;
  v4u_t g0;
  g0[0] = 1u;                                              // count=1, is_restore=0
  g0[1] = lds_off;                                         // lds_addr (bytes)
  g0[2] = (uint32_t)ga;                                    // global_addr[31:0]
  g0[3] = ((uint32_t)(ga >> 32) & 0x01FFFFFFu) | 0x80000000u; // addr[56:32] | type=2
  v8i_t g1;
  g1[0] = (int)((1u << 16)      // data_size = 2B
              | (1u << 20)      // pad_enable
              | (4u << 22)      // pad_interval: 32 DWORDs (128B)
              | (3u << 25));    // pad_amount:   4 DWORDs (16B)
  g1[1] = 64 << 16;             // tensor_dim0 = 64 (elements)
  g1[2] = 64 << 16;             // tensor_dim1 = 64
  g1[3] = 64 << 16;             // tile_dim0   = 64
  g1[4] = 64;                   // tile_dim1   = 64, tile_dim2 = 0
  g1[5] = 64;                   // tensor_dim0_stride = 64
  g1[6] = 0;
  g1[7] = 0;
  v4i_t z4 = {0, 0, 0, 0};
#if __clang_major__ >= 23
  v8i_t z8 = {0, 0, 0, 0, 0, 0, 0, 0};
  __builtin_amdgcn_tensor_load_to_lds(g0, g1, z4, z4, z8, 0);
#else
  __builtin_amdgcn_tensor_load_to_lds(g0, g1, z4, z4, 0);
#endif
#else
  (void)lds_off; (void)src;
#endif
}

// ---- compute ---------------------------------------------------------------

// One 64x64 tanh layer for this wave's 32 rows: hout = tanh(hin @ W + b).
// A-fragments (reused across the 4 N-tiles) are hoisted into registers once
// per M-tile; B-fragments are loaded right before their single WMMA.
__device__ __forceinline__ void layer_wmma(const _Float16* __restrict__ hin,
                                           const _Float16* __restrict__ wt,
                                           const float* __restrict__ bias,
                                           _Float16* __restrict__ hout,
                                           int lane)
{
  const int l15  = lane & 15;
  const int hi   = lane >> 4;
  const int koff = hi * 8;          // lanes<16: K 0-7/16-23 ; lanes>=16: K 8-15/24-31
  #pragma unroll
  for (int mt = 0; mt < 2; ++mt) {
    const _Float16* arow = hin + (mt * 16 + l15) * HPAD;
    AFrag a0, a1;                   // K-chunk 0 (K 0..31) and 1 (K 32..63)
    a0.f4[0] = *(const float4*)(arow + koff);
    a0.f4[1] = *(const float4*)(arow + koff + 16);
    a1.f4[0] = *(const float4*)(arow + 32 + koff);
    a1.f4[1] = *(const float4*)(arow + 32 + koff + 16);
    #pragma unroll
    for (int nt = 0; nt < 4; ++nt) {
      const int col = nt * 16 + l15;
      const _Float16* brow = wt + col * HPAD;
      AFrag b0, b1;
      b0.f4[0] = *(const float4*)(brow + koff);
      b0.f4[1] = *(const float4*)(brow + koff + 16);
      b1.f4[0] = *(const float4*)(brow + 32 + koff);
      b1.f4[1] = *(const float4*)(brow + 32 + koff + 16);
      v8f acc = {0.f, 0.f, 0.f, 0.f, 0.f, 0.f, 0.f, 0.f};
      acc = __builtin_amdgcn_wmma_f32_16x16x32_f16(
              false, a0.v, false, b0.v, (short)0, acc, false, false);
      acc = __builtin_amdgcn_wmma_f32_16x16x32_f16(
              false, a1.v, false, b1.v, (short)0, acc, false, false);
      const float bv    = bias[col];
      const int   rbase = mt * 16 + hi * 8;
      #pragma unroll
      for (int r = 0; r < 8; ++r)
        hout[(rbase + r) * HPAD + col] = (_Float16)fast_tanh(acc[r] + bv);
    }
  }
}

__device__ __forceinline__ void mlp_forward(SMem& s, int lane,
                                            _Float16* hA, _Float16* hB,
                                            float x0, float x1, float x2,
                                            float& o0, float& o1, float& o2)
{
  // layer 1 (d=3 input: per-lane VALU, write own row as f16)
  #pragma unroll
  for (int c = 0; c < 8; ++c) {
    union { float4 f4; _Float16 hh[8]; } pk;
    #pragma unroll
    for (int j = 0; j < 8; ++j) {
      int n = c * 8 + j;
      float a = s.b1[n] + x0 * s.W1f[n] + x1 * s.W1f[64 + n] + x2 * s.W1f[128 + n];
      pk.hh[j] = (_Float16)fast_tanh(a);
    }
    *(float4*)(hA + lane * HPAD + c * 8) = pk.f4;
  }
  layer_wmma(hA, s.W2t, s.b2, hB, lane);
  layer_wmma(hB, s.W3t, s.b3, hA, lane);
  // head (out-dim <= 3): per-lane dot products over own row
  float a0 = s.b4[0], a1 = s.b4[1], a2 = s.b4[2];
  #pragma unroll
  for (int c = 0; c < 8; ++c) {
    union { float4 f4; _Float16 hh[8]; } pk;
    pk.f4 = *(const float4*)(hA + lane * HPAD + c * 8);
    #pragma unroll
    for (int j = 0; j < 8; ++j) {
      float hv = (float)pk.hh[j];
      int   k  = c * 8 + j;
      a0 = __builtin_fmaf(hv, s.W4f[k * 4 + 0], a0);
      a1 = __builtin_fmaf(hv, s.W4f[k * 4 + 1], a1);
      a2 = __builtin_fmaf(hv, s.W4f[k * 4 + 2], a2);
    }
  }
  o0 = a0; o1 = a1; o2 = a2;
}

// ---- kernels ---------------------------------------------------------------

// One-time pre-pass: gW2/gW3 f32 [n][k][ncol] -> f16 transposed compact [n][ncol][k]
__global__ void __launch_bounds__(256) convert_weights(
    const float* __restrict__ gW2, const float* __restrict__ gW3,
    _Float16* __restrict__ w2t, _Float16* __restrict__ w3t, int total)
{
  int i = blockIdx.x * 256 + threadIdx.x;
  if (i >= total) return;
  int n = i >> 12;                      // step
  int r = i & 4095;
  int ncol = r >> 6, k = r & 63;        // write-linear: [ncol][k]
  size_t src = (size_t)n * 4096 + (size_t)k * 64 + ncol;
  w2t[i] = (_Float16)gW2[src];
  w3t[i] = (_Float16)gW3[src];
}

template<int WMODE>   // 0: stage f32 weights from global; 1: f16 pre-converted (+TDM)
__global__ void __launch_bounds__(NTHREADS) bsde_rollout(
    const float* __restrict__ xt0, const float* __restrict__ dBt,
    const float* uW1, const float* ub1, const float* uW2, const float* ub2,
    const float* uW3, const float* ub3, const float* uW4, const float* ub4,
    const float* gW1, const float* gb1, const float* gW2, const float* gb2,
    const float* gW3, const float* gb3, const float* gW4, const float* gb4,
    const _Float16* __restrict__ w2t16, const _Float16* __restrict__ w3t16,
    float* __restrict__ out, int Bsz, int Nsteps)
{
  __shared__ SMem s;
  const int    tid  = threadIdx.x;
  const int    lane = tid & 31;
  const int    wave = tid >> 5;
  const size_t p    = (size_t)blockIdx.x * NTHREADS + tid;

  _Float16* hA = &s.h[0][wave * 32 * HPAD];
  _Float16* hB = &s.h[1][wave * 32 * HPAD];

  float x0 = xt0[p * 3 + 0], x1 = xt0[p * 3 + 1], x2 = xt0[p * 3 + 2];
  float g00 = 1.f, g01 = 0.f, g02 = 0.f;     // transport matrix gt = I
  float g10 = 0.f, g11 = 1.f, g12 = 0.f;
  float g20 = 0.f, g21 = 0.f, g22 = 1.f;

  // u_pre0 = u_mlp(xt0): u-weights staged from f32 in both modes (one-time)
  stage_small(s, tid, uW1, ub1, ub2, ub3, uW4, ub4, /*e_out=*/1);
  stage_big_f32(s, tid, uW2, uW3);
  __syncthreads();
  float u_pre, du1, du2;
  mlp_forward(s, lane, hA, hB, x0, x1, x2, u_pre, du1, du2);

  const float sq2D = 1.0f;   // sqrt(2*D), D = 0.5

  for (int n = 0; n < Nsteps; ++n) {
    __syncthreads();   // all waves done with previous step's weights/activations
    if constexpr (WMODE == 1) {
#if HAVE_TDM
      if (wave == 0) {   // one wave drives the Tensor Data Mover for both tiles
        tdm_load_tile((uint32_t)(uintptr_t)&s.W2t[0], w2t16 + (size_t)n * 4096);
        tdm_load_tile((uint32_t)(uintptr_t)&s.W3t[0], w3t16 + (size_t)n * 4096);
      }
#else
      stage_big_f16(s, tid, w2t16 + (size_t)n * 4096, w3t16 + (size_t)n * 4096);
#endif
      stage_small(s, tid,
                  gW1 + (size_t)n * 192, gb1 + (size_t)n * 64,
                  gb2 + (size_t)n * 64,  gb3 + (size_t)n * 64,
                  gW4 + (size_t)n * 192, gb4 + (size_t)n * 3, /*e_out=*/3);
      if (n + 1 < Nsteps && tid < 64) {   // warm L2 for next step's f16 tiles
        __builtin_prefetch(w2t16 + (size_t)(n + 1) * 4096 + tid * 64, 0, 1);
        __builtin_prefetch(w3t16 + (size_t)(n + 1) * 4096 + tid * 64, 0, 1);
      }
#if HAVE_TDM
      if (wave == 0) __builtin_amdgcn_s_wait_tensorcnt(0);
#endif
    } else {
      stage_big_f32(s, tid, gW2 + (size_t)n * 4096, gW3 + (size_t)n * 4096);
      stage_small(s, tid,
                  gW1 + (size_t)n * 192, gb1 + (size_t)n * 64,
                  gb2 + (size_t)n * 64,  gb3 + (size_t)n * 64,
                  gW4 + (size_t)n * 192, gb4 + (size_t)n * 3, /*e_out=*/3);
      if (n + 1 < Nsteps) {
        __builtin_prefetch(gW2 + (size_t)(n + 1) * 4096 + tid * 32, 0, 1);
        __builtin_prefetch(gW3 + (size_t)(n + 1) * 4096 + tid * 32, 0, 1);
      }
    }
    __syncthreads();   // weights visible to all waves

    const float* db = dBt + ((size_t)n * Bsz + p) * 3;
    float db0 = db[0], db1 = db[1], db2 = db[2];

    float gu0, gu1, gu2;
    mlp_forward(s, lane, hA, hB, x0, x1, x2, gu0, gu1, gu2);

    // gu_t = gu^T @ gt ; u_pre += sq2D * <gu_t, dB>
    float t0 = gu0 * g00 + gu1 * g10 + gu2 * g20;
    float t1 = gu0 * g01 + gu1 * g11 + gu2 * g21;
    float t2 = gu0 * g02 + gu1 * g12 + gu2 * g22;
    u_pre += sq2D * (t0 * db0 + t1 * db1 + t2 * db2);

    // x += gt @ (sq2D * dB)
    float s0 = sq2D * db0, s1 = sq2D * db1, s2 = sq2D * db2;
    x0 += g00 * s0 + g01 * s1 + g02 * s2;
    x1 += g10 * s0 + g11 * s1 + g12 * s2;
    x2 += g20 * s0 + g21 * s1 + g22 * s2;

    // reflect across sphere |x| = R = 1 (branchless: EXEC stays uniform for WMMA)
    float r   = __builtin_amdgcn_sqrtf(x0 * x0 + x1 * x1 + x2 * x2);
    float inv = __builtin_amdgcn_rcpf(fmaxf(r, 1e-12f));
    float nb0 = x0 * inv, nb1 = x1 * inv, nb2 = x2 * inv;
    bool  ob  = r > 1.0f;
    float cc  = 2.0f * (r - 1.0f);
    float rx0 = x0 - cc * nb0, rx1 = x1 - cc * nb1, rx2 = x2 - cc * nb2;
    x0 = ob ? rx0 : x0;  x1 = ob ? rx1 : x1;  x2 = ob ? rx2 : x2;

    // gt := (I - 2 nb nb^T) gt for reflected paths
    float q0 = nb0 * g00 + nb1 * g10 + nb2 * g20;
    float q1 = nb0 * g01 + nb1 * g11 + nb2 * g21;
    float q2 = nb0 * g02 + nb1 * g12 + nb2 * g22;
    float h00 = g00 - 2.f * nb0 * q0, h01 = g01 - 2.f * nb0 * q1, h02 = g02 - 2.f * nb0 * q2;
    float h10 = g10 - 2.f * nb1 * q0, h11 = g11 - 2.f * nb1 * q1, h12 = g12 - 2.f * nb1 * q2;
    float h20 = g20 - 2.f * nb2 * q0, h21 = g21 - 2.f * nb2 * q1, h22 = g22 - 2.f * nb2 * q2;
    g00 = ob ? h00 : g00; g01 = ob ? h01 : g01; g02 = ob ? h02 : g02;
    g10 = ob ? h10 : g10; g11 = ob ? h11 : g11; g12 = ob ? h12 : g12;
    g20 = ob ? h20 : g20; g21 = ob ? h21 : g21; g22 = ob ? h22 : g22;
  }

  out[p]               = u_pre;                          // u_pre [B]
  out[(size_t)Bsz + p] = x0 * x0 + x1 * x1 + x2 * x2;    // u_rel [B]
}

// ---- host ------------------------------------------------------------------

extern "C" void kernel_launch(void* const* d_in, const int* in_sizes, int n_in,
                              void* d_out, int out_size, void* d_ws, size_t ws_size,
                              hipStream_t stream) {
  (void)n_in; (void)out_size;
  const float* xt0 = (const float*)d_in[0];
  const float* dBt = (const float*)d_in[1];
  const float* uW1 = (const float*)d_in[2];
  const float* ub1 = (const float*)d_in[3];
  const float* uW2 = (const float*)d_in[4];
  const float* ub2 = (const float*)d_in[5];
  const float* uW3 = (const float*)d_in[6];
  const float* ub3 = (const float*)d_in[7];
  const float* uW4 = (const float*)d_in[8];
  const float* ub4 = (const float*)d_in[9];
  const float* gW1 = (const float*)d_in[10];
  const float* gb1 = (const float*)d_in[11];
  const float* gW2 = (const float*)d_in[12];
  const float* gb2 = (const float*)d_in[13];
  const float* gW3 = (const float*)d_in[14];
  const float* gb3 = (const float*)d_in[15];
  const float* gW4 = (const float*)d_in[16];
  const float* gb4 = (const float*)d_in[17];

  const int B = in_sizes[0] / 3;          // 65536
  const int N = in_sizes[10] / (3 * 64);  // 256

  dim3 grid(B / NTHREADS), block(NTHREADS);

  const size_t need = (size_t)2 * N * 4096 * sizeof(_Float16);   // 4 MB for N=256
  if (ws_size >= need) {
    _Float16* w2t = (_Float16*)d_ws;
    _Float16* w3t = w2t + (size_t)N * 4096;
    const int total = N * 4096;
    convert_weights<<<dim3((total + 255) / 256), dim3(256), 0, stream>>>(
        gW2, gW3, w2t, w3t, total);
    bsde_rollout<1><<<grid, block, 0, stream>>>(
        xt0, dBt, uW1, ub1, uW2, ub2, uW3, ub3, uW4, ub4,
        gW1, gb1, gW2, gb2, gW3, gb3, gW4, gb4,
        w2t, w3t, (float*)d_out, B, N);
  } else {
    bsde_rollout<0><<<grid, block, 0, stream>>>(
        xt0, dBt, uW1, ub1, uW2, ub2, uW3, ub3, uW4, ub4,
        gW1, gb1, gW2, gb2, gW3, gb3, gW4, gb4,
        nullptr, nullptr, (float*)d_out, B, N);
  }
}
